// EpisodicMemory_80144089743477
// MI455X (gfx1250) — matrix operand order repaired and checked by reading, not compiled, not run on hardware
//
#include <hip/hip_runtime.h>
#include <cstddef>

typedef __attribute__((ext_vector_type(16))) _Float16 v16h;
typedef __attribute__((ext_vector_type(8)))  _Float16 v8h;
typedef __attribute__((ext_vector_type(4)))  _Float16 v4h;
typedef __attribute__((ext_vector_type(8)))  float    v8f;

namespace {
constexpr int kBS = 8;
constexpr int kB  = 4;
constexpr int kM  = 2048;
constexpr int kD  = 128;
constexpr int kN  = 1024;
constexpr float kNegBig = -1e30f;
constexpr float kBudget = 32.0f;
}

__device__ __forceinline__ v8f wmma_f16f32(v16h a, v16h b, v8f c) {
  // D = A(16x32 f16) * B(32x16 f16) + C(16x16 f32)
  return __builtin_amdgcn_wmma_f32_16x16x32_f16(
      /*neg_a=*/false, a, /*neg_b=*/false, b,
      /*c_mod=*/(short)0, c, /*reuse_a=*/false, /*reuse_b=*/false);
}

// A fragment: 16(M) x 32(K) f16 tile, row-major in LDS, row stride `ld` halves.
// Lane layout: row = lane&15; k = (i&7) + 8*(lane>>4) + 16*(i>>3)
__device__ __forceinline__ v16h load_a_frag(const _Float16* tile, int ld) {
  const int lane = threadIdx.x & 31;
  const int hf = lane >> 4, row = lane & 15;
  const _Float16* p = tile + row * ld + 8 * hf;
  v8h lo = *(const v8h*)p;
  v8h hi = *(const v8h*)(p + 16);
  v16h a;
#pragma unroll
  for (int i = 0; i < 8; ++i) { a[i] = lo[i]; a[i + 8] = hi[i]; }
  return a;
}

// B fragment: 32(K) x 16(N) f16, where the SOURCE is stored transposed:
// tileT[j][k] row-major with row stride `ld`. Lane: col j = lane&15,
// k = (i&7) + 8*(lane>>4) + 16*(i>>3)  (same order as A fragment).
__device__ __forceinline__ v16h load_b_frag(const _Float16* tileT, int ld) {
  const int lane = threadIdx.x & 31;
  const int hf = lane >> 4, col = lane & 15;
  const _Float16* p = tileT + col * ld + 8 * hf;
  v8h lo = *(const v8h*)p;
  v8h hi = *(const v8h*)(p + 16);
  v16h b;
#pragma unroll
  for (int i = 0; i < 8; ++i) { b[i] = lo[i]; b[i + 8] = hi[i]; }
  return b;
}

__device__ __forceinline__ float softplusf(float x) { return log1pf(__expf(x)); }
__device__ __forceinline__ float sigmoidf(float x) { return 1.0f / (1.0f + __expf(-x)); }

// ---------------------------------------------------------------------------
// Kernel 1: one attention refinement step (fused flash-attention + gate).
// Block = 128 threads = 4 waves; each wave owns 16 rows of y.
// grid = (N/64, BS). If `sub` != null, writes (y_new - sub)  (y_em fusion).
// ---------------------------------------------------------------------------
__global__ void __launch_bounds__(128)
attn_step_kernel(const float* __restrict__ y_in, float* __restrict__ y_out,
                 const float* __restrict__ sub,
                 const float* __restrict__ emK, const float* __restrict__ emV,
                 const float* __restrict__ emS,
                 const float* __restrict__ w1, const float* __restrict__ w2,
                 const float* __restrict__ gb,
                 const float* __restrict__ raw_tau, const int* __restrict__ bptr) {
  __shared__ __align__(16) _Float16 sK[32 * 128];   // K tile, row-major [m][d]
  __shared__ __align__(16) _Float16 sVT[128 * 32];  // V tile transposed [d][m]
  __shared__ __align__(16) _Float16 sP[4][16 * 32]; // per-wave P tile [n][m]
  __shared__ float sAct[32];

  const int b = bptr[0];
  const int bs = blockIdx.y;
  const int wid = threadIdx.x >> 5;
  const int lane = threadIdx.x & 31;
  const int hf = lane >> 4, colj = lane & 15;
  const int n0 = blockIdx.x * 64 + wid * 16;

  const float tau = softplusf(raw_tau[b]) + 0.1f;
  const float itau = 1.0f / tau;

  // y A-fragments (pre-scaled by 1/tau), 4 chunks of K=32 covering D=128
  v16h yfrag[4];
  {
    const float* yrow = y_in + ((size_t)bs * kN + n0 + colj) * kD;
#pragma unroll
    for (int c = 0; c < 4; ++c) {
      v16h a;
#pragma unroll
      for (int i = 0; i < 8; ++i) {
        a[i]     = (_Float16)(yrow[c * 32 + 8 * hf + i] * itau);
        a[i + 8] = (_Float16)(yrow[c * 32 + 16 + 8 * hf + i] * itau);
      }
      yfrag[c] = a;
    }
  }

  v8f zero = {};
  v8f acc[8];
#pragma unroll
  for (int c = 0; c < 8; ++c) acc[c] = zero;
  float rowmax[8], rowsum[8];
#pragma unroll
  for (int r = 0; r < 8; ++r) { rowmax[r] = kNegBig; rowsum[r] = 0.0f; }

  const float* Kg = emK + (size_t)(bs * kB + b) * kM * kD;
  const float* Vg = emV + (size_t)(bs * kB + b) * kM * kD;
  const float* Sg = emS + (size_t)(bs * kB + b) * kM;

  for (int m0 = 0; m0 < kM; m0 += 32) {
    // prefetch next K/V tiles into cache while this tile computes
    // (one byte/lane covers a cacheline; 128 lanes x 128B = one 16KB tile)
    if (m0 + 32 < kM) {
      __builtin_prefetch(Kg + (size_t)(m0 + 32) * kD + threadIdx.x * 32, 0, 1);
      __builtin_prefetch(Vg + (size_t)(m0 + 32) * kD + threadIdx.x * 32, 0, 1);
    }
    __syncthreads();
    {
      const float4* Kg4 = (const float4*)(Kg + (size_t)m0 * kD);
      const float4* Vg4 = (const float4*)(Vg + (size_t)m0 * kD);
      for (int idx = threadIdx.x; idx < 32 * 32; idx += 128) {
        int mm = idx >> 5, d4 = (idx & 31) * 4;
        float4 kv = Kg4[idx];
        float4 vv = Vg4[idx];
        v4h kh;
        kh[0] = (_Float16)kv.x; kh[1] = (_Float16)kv.y;
        kh[2] = (_Float16)kv.z; kh[3] = (_Float16)kv.w;
        *(v4h*)(sK + mm * 128 + d4) = kh;
        sVT[(d4 + 0) * 32 + mm] = (_Float16)vv.x;
        sVT[(d4 + 1) * 32 + mm] = (_Float16)vv.y;
        sVT[(d4 + 2) * 32 + mm] = (_Float16)vv.z;
        sVT[(d4 + 3) * 32 + mm] = (_Float16)vv.w;
      }
    }
    if (threadIdx.x < 32) sAct[threadIdx.x] = Sg[m0 + threadIdx.x];
    __syncthreads();

    // scores: two 16x16 tiles (m sub-tiles), K reduced over 4 chunks of 32
    v8f sc0 = zero, sc1 = zero;
#pragma unroll
    for (int c = 0; c < 4; ++c) {
      sc0 = wmma_f16f32(yfrag[c], load_b_frag(sK + c * 32, 128), sc0);
      sc1 = wmma_f16f32(yfrag[c], load_b_frag(sK + 16 * 128 + c * 32, 128), sc1);
    }
    const bool act0 = sAct[colj] > 0.0f;
    const bool act1 = sAct[16 + colj] > 0.0f;

    float corr[8];
    _Float16* pb = sP[wid];
#pragma unroll
    for (int r = 0; r < 8; ++r) {
      float s0 = act0 ? sc0[r] : kNegBig;
      float s1 = act1 ? sc1[r] : kNegBig;
      float mx = fmaxf(s0, s1);
#pragma unroll
      for (int k = 1; k < 16; k <<= 1) mx = fmaxf(mx, __shfl_xor(mx, k, 32));
      float nm = fmaxf(rowmax[r], mx);
      float cf = __expf(rowmax[r] - nm);
      rowmax[r] = nm;
      float p0 = __expf(s0 - nm);
      float p1 = __expf(s1 - nm);
      float ps = p0 + p1;
#pragma unroll
      for (int k = 1; k < 16; k <<= 1) ps += __shfl_xor(ps, k, 32);
      rowsum[r] = rowsum[r] * cf + ps;
      corr[r] = cf;
      pb[(r + 8 * hf) * 32 + colj] = (_Float16)p0;
      pb[(r + 8 * hf) * 32 + 16 + colj] = (_Float16)p1;
    }

    // delta accumulation: rescale accumulators, then P(16x32) * V(32x16) per d-chunk
    v16h pA = load_a_frag(pb, 32);
#pragma unroll
    for (int c8 = 0; c8 < 8; ++c8) {
      v8f a = acc[c8];
#pragma unroll
      for (int r = 0; r < 8; ++r) a[r] = a[r] * corr[r];
      acc[c8] = wmma_f16f32(pA, load_b_frag(sVT + c8 * 16 * 32, 32), a);
    }
  }

  // epilogue: normalize, gate, residual
  float inv[8];
#pragma unroll
  for (int r = 0; r < 8; ++r)
    inv[r] = (rowmax[r] < -1e29f) ? 0.0f : 1.0f / rowsum[r];

#pragma unroll
  for (int c8 = 0; c8 < 8; ++c8) {
    int d = c8 * 16 + colj;
    float w1v = w1[b * kD + d], w2v = w2[b * kD + d], gbv = gb[b * kD + d];
#pragma unroll
    for (int r = 0; r < 8; ++r) {
      int n = n0 + r + 8 * hf;
      size_t idx = ((size_t)bs * kN + n) * kD + d;
      float delta = acc[c8][r] * inv[r];
      float yv = y_in[idx];
      float g = sigmoidf(w1v * yv + w2v * delta + gbv);
      float yn = yv + g * delta;
      if (sub) yn -= sub[idx];
      y_out[idx] = yn;
    }
  }
}

// ---------------------------------------------------------------------------
// Kernel 2: per-row logsumexp stats of route scores (w_norm . K / tau_w).
// grid = (N/64, BS). stats[(bs*N+n)*2] = {rowmax, rowsumexp}.
// ---------------------------------------------------------------------------
__global__ void __launch_bounds__(128)
route_stats_kernel(const float* __restrict__ w_cand,
                   const float* __restrict__ emK, const float* __restrict__ emS,
                   const float* __restrict__ raw_tau_w, const int* __restrict__ bptr,
                   float* __restrict__ stats) {
  __shared__ __align__(16) _Float16 sK[32 * 128];
  __shared__ float sAct[32];

  const int b = bptr[0];
  const int bs = blockIdx.y;
  const int wid = threadIdx.x >> 5;
  const int lane = threadIdx.x & 31;
  const int hf = lane >> 4, colj = lane & 15;
  const int n0 = blockIdx.x * 64 + wid * 16;

  const float tw = softplusf(raw_tau_w[b]) + 0.1f;
  const float itw = 1.0f / tw;

  // row inverse norm for this lane's row (lanes L and L+16 share a row)
  const float* wrow = w_cand + ((size_t)bs * kN + n0 + colj) * kD;
  float ss = 0.0f;
  for (int dd = 64 * hf; dd < 64 * hf + 64; ++dd) { float v = wrow[dd]; ss += v * v; }
  ss += __shfl_xor(ss, 16, 32);
  const float scale = itw / fmaxf(sqrtf(ss), 1e-12f);

  v16h wfrag[4];
#pragma unroll
  for (int c = 0; c < 4; ++c) {
    v16h a;
#pragma unroll
    for (int i = 0; i < 8; ++i) {
      a[i]     = (_Float16)(wrow[c * 32 + 8 * hf + i] * scale);
      a[i + 8] = (_Float16)(wrow[c * 32 + 16 + 8 * hf + i] * scale);
    }
    wfrag[c] = a;
  }

  float rowmax[8], rowsum[8];
#pragma unroll
  for (int r = 0; r < 8; ++r) { rowmax[r] = kNegBig; rowsum[r] = 0.0f; }

  const float* Kg = emK + (size_t)(bs * kB + b) * kM * kD;
  const float* Sg = emS + (size_t)(bs * kB + b) * kM;
  v8f zero = {};

  for (int m0 = 0; m0 < kM; m0 += 32) {
    if (m0 + 32 < kM)
      __builtin_prefetch(Kg + (size_t)(m0 + 32) * kD + threadIdx.x * 32, 0, 1);
    __syncthreads();
    {
      const float4* Kg4 = (const float4*)(Kg + (size_t)m0 * kD);
      for (int idx = threadIdx.x; idx < 32 * 32; idx += 128) {
        int mm = idx >> 5, d4 = (idx & 31) * 4;
        float4 kv = Kg4[idx];
        v4h kh;
        kh[0] = (_Float16)kv.x; kh[1] = (_Float16)kv.y;
        kh[2] = (_Float16)kv.z; kh[3] = (_Float16)kv.w;
        *(v4h*)(sK + mm * 128 + d4) = kh;
      }
    }
    if (threadIdx.x < 32) sAct[threadIdx.x] = Sg[m0 + threadIdx.x];
    __syncthreads();

    v8f sc0 = zero, sc1 = zero;
#pragma unroll
    for (int c = 0; c < 4; ++c) {
      sc0 = wmma_f16f32(wfrag[c], load_b_frag(sK + c * 32, 128), sc0);
      sc1 = wmma_f16f32(wfrag[c], load_b_frag(sK + 16 * 128 + c * 32, 128), sc1);
    }
    const bool act0 = sAct[colj] > 0.0f;
    const bool act1 = sAct[16 + colj] > 0.0f;
#pragma unroll
    for (int r = 0; r < 8; ++r) {
      float s0 = act0 ? sc0[r] : kNegBig;
      float s1 = act1 ? sc1[r] : kNegBig;
      float mx = fmaxf(s0, s1);
#pragma unroll
      for (int k = 1; k < 16; k <<= 1) mx = fmaxf(mx, __shfl_xor(mx, k, 32));
      float nm = fmaxf(rowmax[r], mx);
      float cf = __expf(rowmax[r] - nm);
      rowmax[r] = nm;
      float ps = __expf(s0 - nm) + __expf(s1 - nm);
#pragma unroll
      for (int k = 1; k < 16; k <<= 1) ps += __shfl_xor(ps, k, 32);
      rowsum[r] = rowsum[r] * cf + ps;
    }
  }

  if (colj == 0) {
#pragma unroll
    for (int r = 0; r < 8; ++r) {
      int n = n0 + r + 8 * hf;
      stats[((size_t)bs * kN + n) * 2]     = rowmax[r];
      stats[((size_t)bs * kN + n) * 2 + 1] = rowsum[r];
    }
  }
}

// ---------------------------------------------------------------------------
// Kernel 3: transposed routing update. Blocks own 64-slot m-tiles, loop n.
// grid = (M/64, BS). Produces new_K, new_V, new_age, pre-scale new_S.
// ---------------------------------------------------------------------------
__global__ void __launch_bounds__(128)
route_update_kernel(const float* __restrict__ w_cand, const float* __restrict__ novelty,
                    const float* __restrict__ g_em,
                    const float* __restrict__ emK, const float* __restrict__ emV,
                    const float* __restrict__ emS, const float* __restrict__ emAge,
                    const float* __restrict__ raw_tau_w, const int* __restrict__ bptr,
                    const float* __restrict__ stats,
                    float* __restrict__ outK, float* __restrict__ outV,
                    float* __restrict__ outAge, float* __restrict__ wsS) {
  __shared__ __align__(16) _Float16 sWN[32 * 128];   // w_norm row-major [n][d]
  __shared__ __align__(16) _Float16 sWNT[128 * 32];  // w_norm transposed [d][n]
  __shared__ __align__(16) _Float16 sWCT[128 * 32];  // w_cand transposed [d][n]
  __shared__ __align__(16) _Float16 sP[4][16 * 32];  // per-wave P^T [m][n]
  __shared__ float sInv[32];

  const int b = bptr[0];
  const int bs = blockIdx.y;
  const int wid = threadIdx.x >> 5;
  const int lane = threadIdx.x & 31;
  const int hf = lane >> 4, colj = lane & 15;
  const int m0 = blockIdx.x * 64 + wid * 16;

  const float tw = softplusf(raw_tau_w[b]) + 0.1f;
  const float itw = 1.0f / tw;

  const float* Kg = emK + (size_t)(bs * kB + b) * kM * kD;
  const float* Vg = emV + (size_t)(bs * kB + b) * kM * kD;
  const float* Sg = emS + (size_t)(bs * kB + b) * kM;
  const float* Ag = emAge + (size_t)(bs * kB + b) * kM;

  // K A-fragments for this wave's 16 m rows, pre-scaled by 1/tau_w
  v16h kfrag[4];
  {
    const float* krow = Kg + (size_t)(m0 + colj) * kD;
#pragma unroll
    for (int c = 0; c < 4; ++c) {
      v16h a;
#pragma unroll
      for (int i = 0; i < 8; ++i) {
        a[i]     = (_Float16)(krow[c * 32 + 8 * hf + i] * itw);
        a[i + 8] = (_Float16)(krow[c * 32 + 16 + 8 * hf + i] * itw);
      }
      kfrag[c] = a;
    }
  }
  bool act[8];
#pragma unroll
  for (int r = 0; r < 8; ++r) act[r] = Sg[m0 + r + 8 * hf] > 0.0f;

  v8f zero = {};
  v8f accK[8], accV[8];
#pragma unroll
  for (int c = 0; c < 8; ++c) { accK[c] = zero; accV[c] = zero; }
  float colsum[8];
#pragma unroll
  for (int r = 0; r < 8; ++r) colsum[r] = 0.0f;

  for (int nt = 0; nt < kN; nt += 32) {
    if (nt + 32 < kN)
      __builtin_prefetch(w_cand + ((size_t)bs * kN + nt + 32) * kD + threadIdx.x * 32, 0, 1);
    __syncthreads();
    if (threadIdx.x < 32) {
      const float4* wr4 =
          (const float4*)(w_cand + ((size_t)bs * kN + nt + threadIdx.x) * kD);
      float ss = 0.0f;
      for (int q = 0; q < 32; ++q) {
        float4 v = wr4[q];
        ss += v.x * v.x + v.y * v.y + v.z * v.z + v.w * v.w;
      }
      sInv[threadIdx.x] = 1.0f / fmaxf(sqrtf(ss), 1e-12f);
    }
    __syncthreads();
    {
      const float4* Wg4 = (const float4*)(w_cand + ((size_t)bs * kN + nt) * kD);
      for (int idx = threadIdx.x; idx < 32 * 32; idx += 128) {
        int nn = idx >> 5, d4 = (idx & 31) * 4;
        float4 wc4 = Wg4[idx];
        float inv = sInv[nn];
        v4h wnh;
        wnh[0] = (_Float16)(wc4.x * inv); wnh[1] = (_Float16)(wc4.y * inv);
        wnh[2] = (_Float16)(wc4.z * inv); wnh[3] = (_Float16)(wc4.w * inv);
        *(v4h*)(sWN + nn * 128 + d4) = wnh;
        sWNT[(d4 + 0) * 32 + nn] = wnh[0];
        sWNT[(d4 + 1) * 32 + nn] = wnh[1];
        sWNT[(d4 + 2) * 32 + nn] = wnh[2];
        sWNT[(d4 + 3) * 32 + nn] = wnh[3];
        sWCT[(d4 + 0) * 32 + nn] = (_Float16)wc4.x;
        sWCT[(d4 + 1) * 32 + nn] = (_Float16)wc4.y;
        sWCT[(d4 + 2) * 32 + nn] = (_Float16)wc4.z;
        sWCT[(d4 + 3) * 32 + nn] = (_Float16)wc4.w;
      }
    }
    __syncthreads();

    // transposed scores: sc[m][n] = (K/tau_w) . w_norm
    v8f sc0 = zero, sc1 = zero;
#pragma unroll
    for (int c = 0; c < 4; ++c) {
      sc0 = wmma_f16f32(kfrag[c], load_b_frag(sWN + c * 32, 128), sc0);
      sc1 = wmma_f16f32(kfrag[c], load_b_frag(sWN + 16 * 128 + c * 32, 128), sc1);
    }

    int nA = nt + colj, nBi = nt + 16 + colj;
    float mx0 = stats[((size_t)bs * kN + nA) * 2];
    float sm0 = stats[((size_t)bs * kN + nA) * 2 + 1];
    float mx1 = stats[((size_t)bs * kN + nBi) * 2];
    float sm1 = stats[((size_t)bs * kN + nBi) * 2 + 1];
    float nv0 = novelty[(size_t)bs * kN + nA];
    float nv1 = novelty[(size_t)bs * kN + nBi];
    bool fm0 = mx0 < -1e29f, fm1 = mx1 < -1e29f;
    float f0 = fm0 ? 0.0f : nv0 / sm0;
    float f1 = fm1 ? 0.0f : nv1 / sm1;
    float u0 = nv0 * (1.0f / kM), u1 = nv1 * (1.0f / kM);

    _Float16* pb = sP[wid];
#pragma unroll
    for (int r = 0; r < 8; ++r) {
      float p0 = fm0 ? u0 : ((act[r] ? __expf(sc0[r] - mx0) : 0.0f) * f0);
      float p1 = fm1 ? u1 : ((act[r] ? __expf(sc1[r] - mx1) : 0.0f) * f1);
      colsum[r] += p0 + p1;
      pb[(r + 8 * hf) * 32 + colj] = (_Float16)p0;
      pb[(r + 8 * hf) * 32 + 16 + colj] = (_Float16)p1;
    }

    v16h pA = load_a_frag(pb, 32);
#pragma unroll
    for (int c8 = 0; c8 < 8; ++c8) {
      accK[c8] = wmma_f16f32(pA, load_b_frag(sWNT + c8 * 16 * 32, 32), accK[c8]);
      accV[c8] = wmma_f16f32(pA, load_b_frag(sWCT + c8 * 16 * 32, 32), accV[c8]);
    }
  }

  // epilogue: reduce column sums over n, normalize updates, blend outputs
  const float gem = g_em[bs];
  float alpha[8], rdn[8];
#pragma unroll
  for (int r = 0; r < 8; ++r) {
    float cs = colsum[r];
#pragma unroll
    for (int k = 1; k < 16; k <<= 1) cs += __shfl_xor(cs, k, 32);
    rdn[r] = 1.0f / fmaxf(cs, 1e-8f);
    float ragg = cs * (1.0f / kN);
    alpha[r] = fminf(gem * ragg, 1.0f);
  }
  float nsq[8];
#pragma unroll
  for (int r = 0; r < 8; ++r) nsq[r] = 0.0f;
#pragma unroll
  for (int c8 = 0; c8 < 8; ++c8) {
#pragma unroll
    for (int r = 0; r < 8; ++r) {
      float uk = accK[c8][r] * rdn[r];
      accK[c8][r] = uk;
      accV[c8][r] = accV[c8][r] * rdn[r];
      nsq[r] += uk * uk;
    }
  }
  float invu[8];
#pragma unroll
  for (int r = 0; r < 8; ++r) {
    float s = nsq[r];
#pragma unroll
    for (int k = 1; k < 16; k <<= 1) s += __shfl_xor(s, k, 32);
    invu[r] = 1.0f / fmaxf(sqrtf(s), 1e-12f);
  }
#pragma unroll
  for (int c8 = 0; c8 < 8; ++c8) {
    int d = c8 * 16 + colj;
#pragma unroll
    for (int r = 0; r < 8; ++r) {
      int m = m0 + r + 8 * hf;
      size_t ei = ((size_t)(bs * kB + b) * kM + m) * kD + d;
      size_t oi = ((size_t)bs * kM + m) * kD + d;
      float a = alpha[r];
      outK[oi] = (1.0f - a) * emK[ei] + a * accK[c8][r] * invu[r];
      outV[oi] = (1.0f - a) * emV[ei] + a * accV[c8][r];
    }
  }
  if (colj == 0) {
#pragma unroll
    for (int r = 0; r < 8; ++r) {
      int m = m0 + r + 8 * hf;
      float a = alpha[r];
      float sp = fminf(fmaxf(Sg[m] + a, 0.0f), 3.0f);
      wsS[(size_t)bs * kM + m] = sp;
      outAge[(size_t)bs * kM + m] = Ag[m] * (1.0f - a);
    }
  }
}

// ---------------------------------------------------------------------------
// Kernel 4: per-batch budget rescale of new_S. grid = BS blocks x 256.
// ---------------------------------------------------------------------------
__global__ void __launch_bounds__(256)
s_finalize_kernel(const float* __restrict__ wsS, float* __restrict__ outS) {
  __shared__ float red[8];
  const int bs = blockIdx.x;
  float part = 0.0f;
  for (int i = threadIdx.x; i < kM; i += 256) part += wsS[(size_t)bs * kM + i];
#pragma unroll
  for (int k = 1; k < 32; k <<= 1) part += __shfl_xor(part, k, 32);
  if ((threadIdx.x & 31) == 0) red[threadIdx.x >> 5] = part;
  __syncthreads();
  float tot = 0.0f;
#pragma unroll
  for (int w = 0; w < 8; ++w) tot += red[w];
  float scale = fminf(1.0f, kBudget / fmaxf(tot, 1e-8f));
  for (int i = threadIdx.x; i < kM; i += 256)
    outS[(size_t)bs * kM + i] = wsS[(size_t)bs * kM + i] * scale;
}

extern "C" void kernel_launch(void* const* d_in, const int* in_sizes, int n_in,
                              void* d_out, int out_size, void* d_ws, size_t ws_size,
                              hipStream_t stream) {
  (void)in_sizes; (void)n_in; (void)out_size; (void)ws_size;
  const float* seed    = (const float*)d_in[0];
  const float* w_cand  = (const float*)d_in[1];
  const float* novelty = (const float*)d_in[2];
  const float* g_em    = (const float*)d_in[3];
  const float* emK     = (const float*)d_in[4];
  const float* emV     = (const float*)d_in[5];
  const float* emS     = (const float*)d_in[6];
  const float* emAge   = (const float*)d_in[7];
  const float* w1      = (const float*)d_in[8];
  const float* w2      = (const float*)d_in[9];
  const float* gb      = (const float*)d_in[10];
  const float* rtau    = (const float*)d_in[11];
  const float* rtauw   = (const float*)d_in[12];
  const int*   bptr    = (const int*)d_in[13];

  float* out  = (float*)d_out;
  float* yem  = out;                                   // (BS,N,D)
  float* oK   = yem + (size_t)kBS * kN * kD;           // (BS,M,D)
  float* oV   = oK + (size_t)kBS * kM * kD;            // (BS,M,D)
  float* oS   = oV + (size_t)kBS * kM * kD;            // (BS,M)
  float* oAge = oS + (size_t)kBS * kM;                 // (BS,M)

  float* stats = (float*)d_ws;                         // (BS,N,2)
  float* wsS   = stats + (size_t)kBS * kN * 2;         // (BS,M)

  dim3 blk(128);
  dim3 gAttn(kN / 64, kBS);
  // step 1: seed -> yem region (used as y scratch; fully rewritten by step 2)
  attn_step_kernel<<<gAttn, blk, 0, stream>>>(seed, yem, nullptr, emK, emV, emS,
                                              w1, w2, gb, rtau, bptr);
  // step 2 (in place, blocks only touch their own rows) + y_em = y - seed
  attn_step_kernel<<<gAttn, blk, 0, stream>>>(yem, yem, seed, emK, emV, emS,
                                              w1, w2, gb, rtau, bptr);
  route_stats_kernel<<<gAttn, blk, 0, stream>>>(w_cand, emK, emS, rtauw, bptr, stats);
  dim3 gUpd(kM / 64, kBS);
  route_update_kernel<<<gUpd, blk, 0, stream>>>(w_cand, novelty, g_em, emK, emV, emS,
                                                emAge, rtauw, bptr, stats,
                                                oK, oV, oAge, wsS);
  s_finalize_kernel<<<dim3(kBS), dim3(256), 0, stream>>>(wsS, oS);
}
